// MLPCUDA_81604378624118
// MI455X (gfx1250) — matrix operand order repaired and checked by reading, not compiled, not run on hardware
//
#include <hip/hip_runtime.h>

// ---------------------------------------------------------------------------
// Problem constants (from reference)
// ---------------------------------------------------------------------------
#define D_DIM   4096
#define F_DIM   14336
#define T_DIM   10240
#define FT_DIM  4096      // F - T
#define K_TOP   200
#define H_DIM   1024
#define N_TOK   2048      // 4 * 512

// GEMM tiling
#define BM 128
#define BK 32
#define BNG 128           // generic GEMM block-N (wave tile 32x64, 8 WMMA/iter)
#define BNF 64            // fused up/gate block-N (wave tile 32x32, dual acc)
#define LDA 40            // BK + 8 pad (conflict-free b128 LDS reads)
#define LDB 40

typedef __attribute__((ext_vector_type(16))) __bf16 v16bf;
typedef __attribute__((ext_vector_type(8)))  float  v8f;
typedef __attribute__((ext_vector_type(4)))  float  v4f;

union FragU { v16bf v; v4f f[2]; };

__device__ __forceinline__ unsigned short f2bf(float x) {
    unsigned int u = __float_as_uint(x);
    u = (u + 0x7FFFu + ((u >> 16) & 1u)) >> 16;   // RNE
    return (unsigned short)u;
}

// ---------------------------------------------------------------------------
// CDNA5 async global->LDS staging (ASYNCcnt-tracked, bypasses VGPRs)
// Low 32 bits of a flat shared pointer == wave-relative LDS byte offset.
// ---------------------------------------------------------------------------
__device__ __forceinline__ void async_b128(void* lds_dst, const void* gsrc) {
    unsigned int l = (unsigned int)(unsigned long long)lds_dst;
    asm volatile("global_load_async_to_lds_b128 %0, %1, off"
                 :: "v"(l), "v"(gsrc) : "memory");
}
__device__ __forceinline__ void async_wait0() {
    asm volatile("s_wait_asynccnt 0x0" ::: "memory");
}

// ---------------------------------------------------------------------------
// fp32 -> bf16 converter (4 elements / thread)
// ---------------------------------------------------------------------------
__global__ __launch_bounds__(256)
void cvt_bf16_kernel(const float* __restrict__ src, unsigned short* __restrict__ dst, long n4) {
    long i = (long)blockIdx.x * 256 + threadIdx.x;
    if (i < n4) {
        v4f f = *(const v4f*)(src + 4 * i);
        unsigned long long p =
            (unsigned long long)f2bf(f.x)        |
            ((unsigned long long)f2bf(f.y) << 16) |
            ((unsigned long long)f2bf(f.z) << 32) |
            ((unsigned long long)f2bf(f.w) << 48);
        *(unsigned long long*)(dst + 4 * i) = p;
    }
}

// ---------------------------------------------------------------------------
// down_w [F, D] fp32  ->  [D, F] bf16 (LDS-tiled transpose)
// ---------------------------------------------------------------------------
__global__ __launch_bounds__(256)
void transpose_cvt_kernel(const float* __restrict__ src, unsigned short* __restrict__ dst) {
    __shared__ float tile[32][33];
    const int fb = blockIdx.y * 32;
    const int db = blockIdx.x * 32;
    const int tx = threadIdx.x & 31;
    const int ty = threadIdx.x >> 5;      // 0..7
#pragma unroll
    for (int i = 0; i < 4; ++i) {
        int f = fb + ty + i * 8;
        tile[ty + i * 8][tx] = src[(size_t)f * D_DIM + db + tx];
    }
    __syncthreads();
#pragma unroll
    for (int i = 0; i < 4; ++i) {
        int d = db + ty + i * 8;
        dst[(size_t)d * F_DIM + fb + tx] = f2bf(tile[tx][ty + i * 8]);
    }
}

// ---------------------------------------------------------------------------
// Fragment loaders following the CDNA5 16-bit WMMA VGPR layouts (ISA 7.12.2)
// ---------------------------------------------------------------------------
__device__ __forceinline__ v16bf load_a_frag(const unsigned short* __restrict__ As, int lane) {
    const int r  = lane & 15;
    const int kb = (lane >> 4) * 8;
    FragU u;
    u.f[0] = *(const v4f*)(As + r * LDA + kb);
    u.f[1] = *(const v4f*)(As + r * LDA + kb + 16);
    return u.v;
}

__device__ __forceinline__ v16bf load_b_frag(const unsigned short* __restrict__ Bs, int lane) {
    const int n  = lane & 15;
    const int kb = (lane >> 4) * 8;
    FragU u;
    u.f[0] = *(const v4f*)(Bs + n * LDB + kb);
    u.f[1] = *(const v4f*)(Bs + n * LDB + kb + 16);
    return u.v;
}

// ---------------------------------------------------------------------------
// Generic GEMM:  C[M, Nout] = A[M, K] * B[Nout, K]^T   (bf16 in, f32 acc)
// OMODE 0: f32 out;  OMODE 1: bf16 out with relu
// 256 threads = 8 waves; block tile 128x128; wave tile 32x64 (2x4 WMMA tiles)
// ---------------------------------------------------------------------------
template <int OMODE>
__global__ __launch_bounds__(256)
void gemm_bf16_kernel(const unsigned short* __restrict__ A,
                      const unsigned short* __restrict__ B,
                      void* __restrict__ Cout,
                      int Nout, int Kdim) {
    __shared__ unsigned short As[2][BM * LDA];
    __shared__ unsigned short Bs[2][BNG * LDB];

    const int tid  = threadIdx.x;
    const int lane = tid & 31;
    const int wave = tid >> 5;
    const int wm   = wave >> 1;   // 0..3  -> 32-row band
    const int wn   = wave & 1;    // 0..1  -> 64-col band
    const int mBase = blockIdx.y * BM;
    const int nBase = blockIdx.x * BNG;

    v8f acc[2][4] = {};

    auto stage = [&](int buf, int k0) {
#pragma unroll
        for (int i = 0; i < 2; ++i) {
            int idx = tid + i * 256;          // 0..511
            int r = idx >> 2;
            int s = (idx & 3) * 8;
            async_b128(&As[buf][r * LDA + s],
                       &A[(size_t)(mBase + r) * Kdim + k0 + s]);
            async_b128(&Bs[buf][r * LDB + s],
                       &B[(size_t)(nBase + r) * Kdim + k0 + s]);
        }
        // pull the weight panel two k-tiles ahead into GL2
        if (k0 + 2 * BK < Kdim)
            __builtin_prefetch(&B[(size_t)(nBase + (tid >> 1)) * Kdim + k0 + 2 * BK], 0, 0);
    };

    stage(0, 0);
    async_wait0();
    __syncthreads();

    const int nk = Kdim / BK;
    for (int kt = 0; kt < nk; ++kt) {
        const int buf = kt & 1;
        if (kt + 1 < nk) stage(buf ^ 1, (kt + 1) * BK);

        v16bf af[2], bf[4];
#pragma unroll
        for (int ms = 0; ms < 2; ++ms)
            af[ms] = load_a_frag(&As[buf][(wm * 32 + ms * 16) * LDA], lane);
#pragma unroll
        for (int ns = 0; ns < 4; ++ns)
            bf[ns] = load_b_frag(&Bs[buf][(wn * 64 + ns * 16) * LDB], lane);
#pragma unroll
        for (int ms = 0; ms < 2; ++ms)
#pragma unroll
            for (int ns = 0; ns < 4; ++ns)
                acc[ms][ns] = __builtin_amdgcn_wmma_f32_16x16x32_bf16(
                    false, af[ms], false, bf[ns], (short)0, acc[ms][ns], false, false);
        async_wait0();
        __syncthreads();
    }

    // Epilogue: C layout -> VGPR v, lanes 0-15: M=v, lanes 16-31: M=v+8
#pragma unroll
    for (int ms = 0; ms < 2; ++ms)
#pragma unroll
        for (int ns = 0; ns < 4; ++ns) {
            const int col  = nBase + wn * 64 + ns * 16 + (lane & 15);
            const int row0 = mBase + wm * 32 + ms * 16 + (lane >> 4) * 8;
#pragma unroll
            for (int v = 0; v < 8; ++v) {
                const int row = row0 + v;
                float val = acc[ms][ns][v];
                if (OMODE == 1) {
                    val = fmaxf(val, 0.0f);
                    ((unsigned short*)Cout)[(size_t)row * Nout + col] = f2bf(val);
                } else {
                    ((float*)Cout)[(size_t)row * Nout + col] = val;
                }
            }
        }
}

// ---------------------------------------------------------------------------
// Fused up/gate GEMM -> ir[N, F] bf16
//   col < T : relu(up)*relu(gate)
//   col >= T: up*gate*mask[row, col-T]      (mask in {0,1} => (u*m)*(g*m)=u*g*m)
// ---------------------------------------------------------------------------
__global__ __launch_bounds__(256)
void gemm_upgate_kernel(const unsigned short* __restrict__ A,   // xb [N, D]
                        const unsigned short* __restrict__ Bu,  // up_w bf16 [F, D]
                        const unsigned short* __restrict__ Bg,  // gate_w bf16 [F, D]
                        const unsigned short* __restrict__ maskb, // [N, FT]
                        unsigned short* __restrict__ ir,        // [N, F]
                        int Kdim) {
    __shared__ unsigned short As [2][BM * LDA];
    __shared__ unsigned short BsU[2][BNF * LDB];
    __shared__ unsigned short BsG[2][BNF * LDB];

    const int tid  = threadIdx.x;
    const int lane = tid & 31;
    const int wave = tid >> 5;
    const int wm   = wave >> 1;
    const int wn   = wave & 1;
    const int mBase = blockIdx.y * BM;
    const int nBase = blockIdx.x * BNF;

    v8f accU[2][2] = {};
    v8f accG[2][2] = {};

    auto stage = [&](int buf, int k0) {
#pragma unroll
        for (int i = 0; i < 2; ++i) {
            int idx = tid + i * 256;
            int r = idx >> 2;
            int s = (idx & 3) * 8;
            async_b128(&As[buf][r * LDA + s],
                       &A[(size_t)(mBase + r) * Kdim + k0 + s]);
        }
        {
            int r = tid >> 2;
            int s = (tid & 3) * 8;
            async_b128(&BsU[buf][r * LDB + s],
                       &Bu[(size_t)(nBase + r) * Kdim + k0 + s]);
            async_b128(&BsG[buf][r * LDB + s],
                       &Bg[(size_t)(nBase + r) * Kdim + k0 + s]);
        }
        if (k0 + 2 * BK < Kdim) {
            __builtin_prefetch(&Bu[(size_t)(nBase + (tid >> 2)) * Kdim + k0 + 2 * BK], 0, 0);
            __builtin_prefetch(&Bg[(size_t)(nBase + (tid >> 2)) * Kdim + k0 + 2 * BK], 0, 0);
        }
    };

    stage(0, 0);
    async_wait0();
    __syncthreads();

    const int nk = Kdim / BK;
    for (int kt = 0; kt < nk; ++kt) {
        const int buf = kt & 1;
        if (kt + 1 < nk) stage(buf ^ 1, (kt + 1) * BK);

        v16bf af[2], bu[2], bg[2];
#pragma unroll
        for (int ms = 0; ms < 2; ++ms)
            af[ms] = load_a_frag(&As[buf][(wm * 32 + ms * 16) * LDA], lane);
#pragma unroll
        for (int ns = 0; ns < 2; ++ns) {
            bu[ns] = load_b_frag(&BsU[buf][(wn * 32 + ns * 16) * LDB], lane);
            bg[ns] = load_b_frag(&BsG[buf][(wn * 32 + ns * 16) * LDB], lane);
        }
#pragma unroll
        for (int ms = 0; ms < 2; ++ms)
#pragma unroll
            for (int ns = 0; ns < 2; ++ns) {
                accU[ms][ns] = __builtin_amdgcn_wmma_f32_16x16x32_bf16(
                    false, af[ms], false, bu[ns], (short)0, accU[ms][ns], false, false);
                accG[ms][ns] = __builtin_amdgcn_wmma_f32_16x16x32_bf16(
                    false, af[ms], false, bg[ns], (short)0, accG[ms][ns], false, false);
            }
        async_wait0();
        __syncthreads();
    }

#pragma unroll
    for (int ms = 0; ms < 2; ++ms)
#pragma unroll
        for (int ns = 0; ns < 2; ++ns) {
            const int col  = nBase + wn * 32 + ns * 16 + (lane & 15);
            const int row0 = mBase + wm * 32 + ms * 16 + (lane >> 4) * 8;
#pragma unroll
            for (int v = 0; v < 8; ++v) {
                const int row = row0 + v;
                const float uv = accU[ms][ns][v];
                const float gv = accG[ms][ns][v];
                float r;
                if (col < T_DIM) {
                    r = fmaxf(uv, 0.0f) * fmaxf(gv, 0.0f);
                } else {
                    const unsigned short mb = maskb[(size_t)row * FT_DIM + (col - T_DIM)];
                    r = mb ? uv * gv : 0.0f;
                }
                ir[(size_t)row * F_DIM + col] = f2bf(r);
            }
        }
}

// ---------------------------------------------------------------------------
// Exact per-row top-K via 32-bit radix threshold search (ties by index order)
// ---------------------------------------------------------------------------
__global__ __launch_bounds__(256)
void topk_mask_kernel(const float* __restrict__ scores, unsigned short* __restrict__ mask) {
    const int row = blockIdx.x;
    const int tid = threadIdx.x;
    const float* srow = scores + (size_t)row * FT_DIM;

    unsigned int keys[16];
#pragma unroll
    for (int i = 0; i < 16; ++i) {
        unsigned int u = __float_as_uint(srow[tid * 16 + i]);
        keys[i] = u ^ ((u & 0x80000000u) ? 0xFFFFFFFFu : 0x80000000u); // order-preserving
    }

    __shared__ int sred[256];

    unsigned int thr = 0;
    for (int bit = 31; bit >= 0; --bit) {
        const unsigned int cand = thr | (1u << bit);
        int c = 0;
#pragma unroll
        for (int i = 0; i < 16; ++i) c += (keys[i] >= cand) ? 1 : 0;
        __syncthreads();
        sred[tid] = c;
        __syncthreads();
        for (int off = 128; off > 0; off >>= 1) {
            if (tid < off) sred[tid] += sred[tid + off];
            __syncthreads();
        }
        if (sred[0] >= K_TOP) thr = cand;   // largest t with count(key >= t) >= K
    }

    int cg = 0, ceq = 0;
#pragma unroll
    for (int i = 0; i < 16; ++i) {
        cg  += (keys[i] > thr)  ? 1 : 0;
        ceq += (keys[i] == thr) ? 1 : 0;
    }
    __syncthreads();
    sred[tid] = cg;
    __syncthreads();
    for (int off = 128; off > 0; off >>= 1) {
        if (tid < off) sred[tid] += sred[tid + off];
        __syncthreads();
    }
    const int cgTotal = sred[0];

    __syncthreads();
    sred[tid] = ceq;
    __syncthreads();
    for (int off = 1; off < 256; off <<= 1) {          // Hillis-Steele inclusive scan
        int v = (tid >= off) ? sred[tid - off] : 0;
        __syncthreads();
        sred[tid] += v;
        __syncthreads();
    }
    const int eqBefore = sred[tid] - ceq;
    const int quota = K_TOP - cgTotal;

    unsigned short* mrow = mask + (size_t)row * FT_DIM;
    int rank = eqBefore;
#pragma unroll
    for (int i = 0; i < 16; ++i) {
        unsigned short mv = 0;
        if (keys[i] > thr) mv = 0x3F80u;                       // bf16 1.0
        else if (keys[i] == thr && (rank++ < quota)) mv = 0x3F80u;
        mrow[tid * 16 + i] = mv;
    }
}

// ---------------------------------------------------------------------------
// Launcher
// ---------------------------------------------------------------------------
extern "C" void kernel_launch(void* const* d_in, const int* in_sizes, int n_in,
                              void* d_out, int out_size, void* d_ws, size_t ws_size,
                              hipStream_t stream) {
    (void)in_sizes; (void)n_in; (void)out_size; (void)ws_size;

    const float* x      = (const float*)d_in[0];
    const float* bn     = (const float*)d_in[1];
    const float* up_w   = (const float*)d_in[2];
    const float* gate_w = (const float*)d_in[3];
    const float* down_w = (const float*)d_in[4];
    const float* fc1_w  = (const float*)d_in[5];
    const float* fc2_w  = (const float*)d_in[6];

    char* base = (char*)d_ws;
    size_t off = 0;
    auto alloc = [&](size_t bytes) -> void* {
        void* r = base + off;
        off = (off + bytes + 255) & ~(size_t)255;
        return r;
    };
    unsigned short* xb     = (unsigned short*)alloc((size_t)N_TOK  * D_DIM  * 2);
    unsigned short* bnb    = (unsigned short*)alloc((size_t)N_TOK  * D_DIM  * 2);
    unsigned short* upwb   = (unsigned short*)alloc((size_t)F_DIM  * D_DIM  * 2);
    unsigned short* gatewb = (unsigned short*)alloc((size_t)F_DIM  * D_DIM  * 2);
    unsigned short* downtb = (unsigned short*)alloc((size_t)D_DIM  * F_DIM  * 2);
    unsigned short* fc1b   = (unsigned short*)alloc((size_t)H_DIM  * D_DIM  * 2);
    unsigned short* fc2tb  = (unsigned short*)alloc((size_t)FT_DIM * H_DIM  * 2);
    unsigned short* hb     = (unsigned short*)alloc((size_t)N_TOK  * H_DIM  * 2);
    float*          scores = (float*)         alloc((size_t)N_TOK  * FT_DIM * 4);
    unsigned short* maskb  = (unsigned short*)alloc((size_t)N_TOK  * FT_DIM * 2);
    unsigned short* irb    = (unsigned short*)alloc((size_t)N_TOK  * F_DIM  * 2);

    auto cvt = [&](const float* s, unsigned short* d, size_t n) {
        long n4 = (long)(n / 4);
        int blocks = (int)((n4 + 255) / 256);
        cvt_bf16_kernel<<<dim3(blocks), dim3(256), 0, stream>>>(s, d, n4);
    };

    // 1) Convert inputs/weights to bf16
    cvt(x,      xb,     (size_t)N_TOK * D_DIM);
    cvt(bn,     bnb,    (size_t)N_TOK * D_DIM);
    cvt(up_w,   upwb,   (size_t)F_DIM * D_DIM);
    cvt(gate_w, gatewb, (size_t)F_DIM * D_DIM);
    cvt(fc1_w,  fc1b,   (size_t)H_DIM * D_DIM);
    cvt(fc2_w + (size_t)T_DIM * H_DIM, fc2tb, (size_t)FT_DIM * H_DIM);

    // down_w [F,D] -> [D,F] bf16 so the down GEMM has contiguous-K B rows
    transpose_cvt_kernel<<<dim3(D_DIM / 32, F_DIM / 32), dim3(256), 0, stream>>>(down_w, downtb);

    // 2) Predictor: h = relu(bn @ fc1^T)   [2048 x 1024]
    gemm_bf16_kernel<1><<<dim3(H_DIM / BNG, N_TOK / BM), dim3(256), 0, stream>>>(
        bnb, fc1b, (void*)hb, H_DIM, D_DIM);

    // 3) scores = h @ fc2_tail^T           [2048 x 4096] f32
    gemm_bf16_kernel<0><<<dim3(FT_DIM / BNG, N_TOK / BM), dim3(256), 0, stream>>>(
        hb, fc2tb, (void*)scores, FT_DIM, H_DIM);

    // 4) per-row top-200 -> bf16 0/1 mask
    topk_mask_kernel<<<dim3(N_TOK), dim3(256), 0, stream>>>(scores, maskb);

    // 5) fused up/gate + activation/mask -> ir [2048 x 14336] bf16
    gemm_upgate_kernel<<<dim3(F_DIM / BNF, N_TOK / BM), dim3(256), 0, stream>>>(
        xb, upwb, gatewb, maskb, irb, D_DIM);

    // 6) out = ir @ down_w                 [2048 x 4096] f32
    gemm_bf16_kernel<0><<<dim3(D_DIM / BNG, N_TOK / BM), dim3(256), 0, stream>>>(
        irb, downtb, d_out, D_DIM, F_DIM);
}
